// GCN_72456098284043
// MI455X (gfx1250) — matrix operand order repaired and checked by reading, not compiled, not run on hardware
//
#include <hip/hip_runtime.h>

// ---------------------------------------------------------------------------
// Types for CDNA5 WMMA (wave32, 16x16x32 bf16 -> f32)
// ---------------------------------------------------------------------------
typedef __attribute__((ext_vector_type(16))) __bf16 v16bf;
typedef __attribute__((ext_vector_type(8)))  __bf16 v8bf;
typedef __attribute__((ext_vector_type(8)))  float  v8f;

union BFV { v16bf v; v8bf h[2]; };

#define HDIM 256

// A fragment (16x32 bf16): lane L: m = L&15, hh = L>>4.
// elements 0..7  <- K = kbase + 8*hh      .. +7
// elements 8..15 <- K = kbase + 16 + 8*hh .. +7
__device__ __forceinline__ v16bf ld_a_frag(const __bf16* slice, int m, int kbase, int hh) {
  BFV u;
  u.h[0] = *(const v8bf*)(slice + m * HDIM + kbase + 8 * hh);
  u.h[1] = *(const v8bf*)(slice + m * HDIM + kbase + 16 + 8 * hh);
  return u.v;
}

// B fragment (32x16 bf16) from W^T stored [N][Kpad] row-major:
// lane L: n = provided, kh = L>>4; elements 0..7 <- K = kbase + 8*kh .. +7, 8..15 <- K = kbase + 16 + 8*kh .. +7
__device__ __forceinline__ v16bf ld_b_frag(const __bf16* __restrict__ wt, int kpad,
                                           int n, int kbase, int kh) {
  const __bf16* p = wt + (size_t)n * kpad + kbase + 8 * kh;
  BFV u;
  u.h[0] = *(const v8bf*)(p);
  u.h[1] = *(const v8bf*)(p + 16);
  return u.v;
}

// ---------------------------------------------------------------------------
// Fused GIN MLP:  hout = relu( relu(hin*W1 + b1) * W2 + b2 )
// hin: M x K1 fp32 (row stride K1), W1T: [256][K1] bf16, W2T: [256][256] bf16
// Each wave handles one 16-row strip; LDS slice is wave-private (no barriers).
// ---------------------------------------------------------------------------
template <int K1>
__global__ __launch_bounds__(256)
void k_gin_mlp(const float* __restrict__ hin,
               const __bf16* __restrict__ w1t, const float* __restrict__ b1,
               const __bf16* __restrict__ w2t, const float* __restrict__ b2,
               float* __restrict__ hout, int ntile)
{
  __shared__ __bf16 sm[8 * 16 * HDIM];           // 64 KB, 8 KB per wave
  const int wave = threadIdx.x >> 5;             // wave32
  const int lane = threadIdx.x & 31;
  const int tile = blockIdx.x * 8 + wave;
  if (tile >= ntile) return;                     // wave-uniform exit (EXEC stays full)

  __bf16* slice = sm + wave * 16 * HDIM;
  const int rowbase = tile * 16;
  const int m  = lane & 15;                      // row (A) / col (B/C) index
  const int hh = lane >> 4;

  // ---- stage 16 x K1 fp32 -> bf16 in LDS (row stride HDIM) ----
  {
    constexpr int C4 = K1 / 4;
    #pragma unroll
    for (int i = lane; i < 16 * C4; i += 32) {
      const int r = i / C4, c4 = i - r * C4;
      const float4 v = *(const float4*)(hin + (size_t)(rowbase + r) * K1 + c4 * 4);
      __bf16* d = slice + r * HDIM + c4 * 4;
      d[0] = (__bf16)v.x; d[1] = (__bf16)v.y; d[2] = (__bf16)v.z; d[3] = (__bf16)v.w;
    }
  }

  constexpr int KT1 = K1 / 32;
  v16bf af[KT1];
  #pragma unroll
  for (int kt = 0; kt < KT1; ++kt) af[kt] = ld_a_frag(slice, m, kt * 32, hh);

  // ---- GEMM1 + bias + relu -> t (bf16, back into same LDS slice) ----
  #pragma unroll
  for (int nt = 0; nt < 16; nt += 2) {
    const float bv0 = b1[nt * 16 + m];
    const float bv1 = b1[(nt + 1) * 16 + m];
    v8f acc0, acc1;
    #pragma unroll
    for (int r = 0; r < 8; ++r) { acc0[r] = bv0; acc1[r] = bv1; }
    #pragma unroll
    for (int kt = 0; kt < KT1; ++kt) {
      const v16bf bf0 = ld_b_frag(w1t, K1, nt * 16 + m, kt * 32, hh);
      const v16bf bf1 = ld_b_frag(w1t, K1, (nt + 1) * 16 + m, kt * 32, hh);
      acc0 = __builtin_amdgcn_wmma_f32_16x16x32_bf16(false, af[kt], false, bf0,
                                                     (short)0, acc0, false, false);
      acc1 = __builtin_amdgcn_wmma_f32_16x16x32_bf16(false, af[kt], false, bf1,
                                                     (short)0, acc1, false, false);
    }
    #pragma unroll
    for (int r = 0; r < 8; ++r) {               // C layout: row = r + 8*hh, col = m
      slice[(r + 8 * hh) * HDIM + nt * 16 + m]       = (__bf16)fmaxf(acc0[r], 0.0f);
      slice[(r + 8 * hh) * HDIM + (nt + 1) * 16 + m] = (__bf16)fmaxf(acc1[r], 0.0f);
    }
  }

  // ---- GEMM2 + bias + relu -> global fp32 ----
  v16bf a2[8];
  #pragma unroll
  for (int kt = 0; kt < 8; ++kt) a2[kt] = ld_a_frag(slice, m, kt * 32, hh);

  float* obase = hout + (size_t)rowbase * HDIM;
  #pragma unroll
  for (int nt = 0; nt < 16; nt += 2) {
    const float bv0 = b2[nt * 16 + m];
    const float bv1 = b2[(nt + 1) * 16 + m];
    v8f acc0, acc1;
    #pragma unroll
    for (int r = 0; r < 8; ++r) { acc0[r] = bv0; acc1[r] = bv1; }
    #pragma unroll
    for (int kt = 0; kt < 8; ++kt) {
      const v16bf bf0 = ld_b_frag(w2t, 256, nt * 16 + m, kt * 32, hh);
      const v16bf bf1 = ld_b_frag(w2t, 256, (nt + 1) * 16 + m, kt * 32, hh);
      acc0 = __builtin_amdgcn_wmma_f32_16x16x32_bf16(false, a2[kt], false, bf0,
                                                     (short)0, acc0, false, false);
      acc1 = __builtin_amdgcn_wmma_f32_16x16x32_bf16(false, a2[kt], false, bf1,
                                                     (short)0, acc1, false, false);
    }
    #pragma unroll
    for (int r = 0; r < 8; ++r) {
      obase[(size_t)(r + 8 * hh) * HDIM + nt * 16 + m]       = fmaxf(acc0[r], 0.0f);
      obase[(size_t)(r + 8 * hh) * HDIM + (nt + 1) * 16 + m] = fmaxf(acc1[r], 0.0f);
    }
  }
}

// ---------------------------------------------------------------------------
// Helper kernels
// ---------------------------------------------------------------------------

// W (K x N fp32, row-major) -> WT (N x Kpad bf16), zero-padded for k >= K
__global__ void k_wconv(const float* __restrict__ w, __bf16* __restrict__ wt,
                        int K, int N, int Kpad)
{
  const int id = blockIdx.x * blockDim.x + threadIdx.x;
  if (id >= N * Kpad) return;
  const int n = id / Kpad, k = id - n * Kpad;
  wt[id] = (k < K) ? (__bf16)w[(size_t)k * N + n] : (__bf16)0.0f;
}

// AGG0[n][0..31] = pad(x[n][0..10])
__global__ void k_agg0_init(const float* __restrict__ x, float* __restrict__ agg, int nn)
{
  const int id = blockIdx.x * blockDim.x + threadIdx.x;
  if (id >= nn * 32) return;
  const int n = id >> 5, c = id & 31;
  agg[id] = (c < 11) ? x[(size_t)n * 11 + c] : 0.0f;
}

// agg[dst[e]][c] += x[src[e]][c], 11 features
__global__ void k_scatter11(const float* __restrict__ x, const int* __restrict__ src,
                            const int* __restrict__ dst, float* __restrict__ agg, int ne)
{
  const long long id = (long long)blockIdx.x * blockDim.x + threadIdx.x;
  if (id >= (long long)ne * 16) return;
  const int e = (int)(id >> 4), c = (int)(id & 15);
  if (c >= 11) return;
  atomicAdd(&agg[(size_t)dst[e] * 32 + c], x[(size_t)src[e] * 11 + c]);
}

// hd[dst[e]][:] += hs[src[e]][:], 256 features (64 consecutive lanes per edge)
__global__ void k_scatter256(const float* __restrict__ hs, const int* __restrict__ src,
                             const int* __restrict__ dst, float* __restrict__ hd, int ne)
{
  const long long id = (long long)blockIdx.x * blockDim.x + threadIdx.x;
  if (id >= (long long)ne * 64) return;
  const int e = (int)(id >> 6), j = (int)(id & 63);
  const float4 v = *(const float4*)(hs + (size_t)src[e] * HDIM + j * 4);
  float* o = hd + (size_t)dst[e] * HDIM + j * 4;
  atomicAdd(o + 0, v.x); atomicAdd(o + 1, v.y);
  atomicAdd(o + 2, v.z); atomicAdd(o + 3, v.w);
}

__global__ void k_pool(const float* __restrict__ h, const int* __restrict__ batch,
                       float* __restrict__ pool, int nn)
{
  const long long id = (long long)blockIdx.x * blockDim.x + threadIdx.x;
  if (id >= (long long)nn * 64) return;
  const int n = (int)(id >> 6), j = (int)(id & 63);
  const int g = batch[n];
  const float4 v = *(const float4*)(h + (size_t)n * HDIM + j * 4);
  float* o = pool + (size_t)g * HDIM + j * 4;
  atomicAdd(o + 0, v.x); atomicAdd(o + 1, v.y);
  atomicAdd(o + 2, v.z); atomicAdd(o + 3, v.w);
}

__global__ void k_count(const int* __restrict__ batch, float* __restrict__ cnt, int nn)
{
  const int id = blockIdx.x * blockDim.x + threadIdx.x;
  if (id >= nn) return;
  atomicAdd(&cnt[batch[id]], 1.0f);
}

// head: out[g] = (mean_pool[g] @ W1 + b1) @ W2 + b2   (no relu)
__global__ __launch_bounds__(256)
void k_head(const float* __restrict__ pool, const float* __restrict__ cnt,
            const float* __restrict__ w1, const float* __restrict__ b1,
            const float* __restrict__ w2, const float* __restrict__ b2,
            float* __restrict__ out)
{
  __shared__ float ps[256];
  __shared__ float red[256];
  const int g = blockIdx.x, t = threadIdx.x;
  const float inv = 1.0f / fmaxf(cnt[g], 1.0f);
  ps[t] = pool[(size_t)g * HDIM + t] * inv;
  __syncthreads();
  float y = b1[t];
  #pragma unroll 8
  for (int k = 0; k < 256; ++k) y = fmaf(ps[k], w1[(size_t)k * 256 + t], y);
  red[t] = y * w2[t];
  __syncthreads();
  for (int s = 128; s > 0; s >>= 1) {
    if (t < s) red[t] += red[t + s];
    __syncthreads();
  }
  if (t == 0) out[g] = red[0] + b2[0];
}

// ---------------------------------------------------------------------------
// Launch
// ---------------------------------------------------------------------------
extern "C" void kernel_launch(void* const* d_in, const int* in_sizes, int n_in,
                              void* d_out, int out_size, void* d_ws, size_t ws_size,
                              hipStream_t stream)
{
  const float* x    = (const float*)d_in[0];
  const int*   ei   = (const int*)d_in[1];    // int32 (JAX demotes int64 w/o x64)
  const int*   bat  = (const int*)d_in[2];
  const float* g0w1 = (const float*)d_in[3];
  const float* g0b1 = (const float*)d_in[4];
  const float* g0w2 = (const float*)d_in[5];
  const float* g0b2 = (const float*)d_in[6];
  const float* gw1  = (const float*)d_in[7];  // (4,256,256)
  const float* gb1  = (const float*)d_in[8];  // (4,256)
  const float* gw2  = (const float*)d_in[9];
  const float* gb2  = (const float*)d_in[10];
  const float* hw1  = (const float*)d_in[11];
  const float* hb1  = (const float*)d_in[12];
  const float* hw2  = (const float*)d_in[13];
  const float* hb2  = (const float*)d_in[14];
  float* out = (float*)d_out;

  const int NN = in_sizes[0] / 11;   // 100000
  const int NE = in_sizes[1] / 2;    // 400000
  const int NG = out_size;           // 5000
  const int NTILE = NN / 16;         // 6250

  // workspace carve (256B aligned)
  char* p = (char*)d_ws;
  auto carve = [&](size_t bytes) {
    char* r = p;
    p += (bytes + 255) & ~(size_t)255;
    return r;
  };
  float*  H0   = (float*)carve((size_t)NN * HDIM * sizeof(float));
  float*  H1   = (float*)carve((size_t)NN * HDIM * sizeof(float));
  float*  AGG0 = (float*)carve((size_t)NN * 32 * sizeof(float));
  float*  POOL = (float*)carve((size_t)NG * HDIM * sizeof(float));
  float*  CNT  = (float*)carve((size_t)NG * sizeof(float));
  __bf16* W1T0 = (__bf16*)carve(256 * 32 * sizeof(__bf16));
  __bf16* W2T0 = (__bf16*)carve(256 * 256 * sizeof(__bf16));
  __bf16* W1T  = (__bf16*)carve(4 * 256 * 256 * sizeof(__bf16));
  __bf16* W2T  = (__bf16*)carve(4 * 256 * 256 * sizeof(__bf16));

  const int* src = ei;
  const int* dst = ei + NE;

  // ---- weights -> transposed bf16 ----
  k_wconv<<<(256 * 32 + 255) / 256, 256, 0, stream>>>(g0w1, W1T0, 11, 256, 32);
  k_wconv<<<(256 * 256 + 255) / 256, 256, 0, stream>>>(g0w2, W2T0, 256, 256, 256);
  for (int i = 0; i < 4; ++i) {
    k_wconv<<<(256 * 256 + 255) / 256, 256, 0, stream>>>(gw1 + (size_t)i * 65536,
                                                         W1T + (size_t)i * 65536, 256, 256, 256);
    k_wconv<<<(256 * 256 + 255) / 256, 256, 0, stream>>>(gw2 + (size_t)i * 65536,
                                                         W2T + (size_t)i * 65536, 256, 256, 256);
  }

  const int mlpGrid = (NTILE + 7) / 8;

  // ---- layer 0 (K=11 padded to 32) ----
  k_agg0_init<<<(NN * 32 + 255) / 256, 256, 0, stream>>>(x, AGG0, NN);
  k_scatter11<<<(int)(((long long)NE * 16 + 255) / 256), 256, 0, stream>>>(x, src, dst, AGG0, NE);
  k_gin_mlp<32><<<mlpGrid, 256, 0, stream>>>(AGG0, W1T0, g0b1, W2T0, g0b2, H0, NTILE);

  // ---- layers 1..4 ----
  for (int i = 0; i < 4; ++i) {
    hipMemcpyAsync(H1, H0, (size_t)NN * HDIM * sizeof(float),
                   hipMemcpyDeviceToDevice, stream);
    k_scatter256<<<(int)(((long long)NE * 64 + 255) / 256), 256, 0, stream>>>(H0, src, dst, H1, NE);
    k_gin_mlp<256><<<mlpGrid, 256, 0, stream>>>(H1, W1T + (size_t)i * 65536, gb1 + (size_t)i * 256,
                                                W2T + (size_t)i * 65536, gb2 + (size_t)i * 256,
                                                H0, NTILE);
  }

  // ---- mean pool + head ----
  hipMemsetAsync(POOL, 0, (size_t)NG * HDIM * sizeof(float), stream);
  hipMemsetAsync(CNT, 0, (size_t)NG * sizeof(float), stream);
  k_pool<<<(int)(((long long)NN * 64 + 255) / 256), 256, 0, stream>>>(H0, bat, POOL, NN);
  k_count<<<(NN + 255) / 256, 256, 0, stream>>>(bat, CNT, NN);
  k_head<<<NG, 256, 0, stream>>>(POOL, CNT, hw1, hb1, hw2, hb2, out);
}
